// VQModel_38766374813930
// MI455X (gfx1250) — compile-verified, matrix operand order checked by reading, not run on hardware
//
#include <hip/hip_runtime.h>
#include <cmath>
#include <vector>

// ============================================================================
// CDNA5 (gfx1250) VQ-VAE forward.
// Strategy: fp16 NHWC activations, implicit-GEMM convs on v_wmma_f32_16x16x32_f16,
// LDS-tiled 128x64 block tiles, 8 wave32 waves (4x2), 32x32 accum per wave.
// ============================================================================

typedef __attribute__((ext_vector_type(16))) _Float16 v16h;
typedef __attribute__((ext_vector_type(8)))  float    v8f;

// ---------------------------------------------------------------------------
// Shared WMMA micro-kernel: consumes one K=32 step from LDS tiles.
// A tile: [128 rows][32 k] (row-major, stride 40), B tile: [32 k][64 n] (stride 72).
// Lane->element mapping per CDNA5 ISA 7.12.2 (16-bit A 16x32 layout).
// ---------------------------------------------------------------------------
__device__ __forceinline__ void mma_step(const _Float16 (*As)[40],
                                         const _Float16 (*Bs)[72],
                                         v8f acc[2][2], int wm, int wn, int lane) {
  const int half = lane >> 4;      // 0: K 0-7/16-23 , 1: K 8-15/24-31
  const int mr   = lane & 15;
  union Frag { v16h v; _Float16 s[16]; };
  Frag a[2], b[2];
#pragma unroll
  for (int t = 0; t < 2; ++t) {
    const int arow = wm * 32 + t * 16 + mr;
    const int bcol = wn * 32 + t * 16 + mr;
#pragma unroll
    for (int e = 0; e < 16; ++e) {
      const int kk = ((e & 8) ? 16 : 0) + (half ? 8 : 0) + (e & 7);
      a[t].s[e] = As[arow][kk];
      b[t].s[e] = Bs[kk][bcol];
    }
  }
#pragma unroll
  for (int i = 0; i < 2; ++i)
#pragma unroll
    for (int j = 0; j < 2; ++j)
      acc[i][j] = __builtin_amdgcn_wmma_f32_16x16x32_f16(
          false, a[i].v, false, b[j].v, (short)0, acc[i][j], false, false);
}

// ---------------------------------------------------------------------------
// Implicit-GEMM conv: out[NHWC] = act[NHWC] (*) w  (+bias)(+residual), fp16 out.
// Weights pre-packed as [K = ky*kw*Cin + kx*Cin + ci][Cout] fp16.
// Handles k=1 and k=3, any stride/pad (asymmetric right/bottom pad falls out of
// the bounds check, matching torch's pad(0,1,0,1)+s2p0 downsample).
// Grid: (M/128, ceil(Cout/64)), block 256 (8 waves).
// ---------------------------------------------------------------------------
__global__ __launch_bounds__(256) void wmma_conv_kernel(
    const _Float16* __restrict__ act, const _Float16* __restrict__ wp,
    const float* __restrict__ bias, const _Float16* __restrict__ resid,
    _Float16* __restrict__ out,
    int H, int W, int Cin, int Ho, int Wo, int Cout, int ksz, int stride, int pad) {
  __shared__ _Float16 As[128][40];
  __shared__ _Float16 Bs[32][72];
  const int Ktot = ksz * ksz * Cin;
  const int tid = threadIdx.x, lane = tid & 31, wave = tid >> 5;
  const int wm = wave >> 1, wn = wave & 1;
  const long long m0 = (long long)blockIdx.x * 128;
  const int n0 = blockIdx.y * 64;

  v8f acc[2][2];
#pragma unroll
  for (int i = 0; i < 2; ++i)
#pragma unroll
    for (int j = 0; j < 2; ++j)
#pragma unroll
      for (int r = 0; r < 8; ++r) acc[i][j][r] = 0.f;

  for (int k0 = 0; k0 < Ktot; k0 += 32) {
    // ---- stage A (im2col on the fly) ----
    for (int idx = tid; idx < 128 * 32; idx += 256) {
      const int row = idx >> 5, kk = idx & 31;
      const long long m = m0 + row;
      const int kg = k0 + kk;
      _Float16 v = (_Float16)0.f;
      if (kg < Ktot) {
        const int ciX = kg % Cin;
        const int r = kg / Cin;
        const int kx = r % ksz, ky = r / ksz;
        const int ow = (int)(m % Wo);
        const long long t = m / Wo;
        const int oh = (int)(t % Ho);
        const int nb = (int)(t / Ho);
        const int ih = oh * stride + ky - pad;
        const int iw = ow * stride + kx - pad;
        if ((unsigned)ih < (unsigned)H && (unsigned)iw < (unsigned)W)
          v = act[(((long long)nb * H + ih) * W + iw) * Cin + ciX];
      }
      As[row][kk] = v;
    }
    // ---- stage B (packed weights) ----
    for (int idx = tid; idx < 32 * 64; idx += 256) {
      const int kk = idx >> 6, nn = idx & 63;
      const int kg = k0 + kk, ng = n0 + nn;
      Bs[kk][nn] = (kg < Ktot && ng < Cout) ? wp[(long long)kg * Cout + ng]
                                            : (_Float16)0.f;
    }
    __syncthreads();
    if (k0 + 32 < Ktot)  // prefetch next weight tile (global_prefetch_b8)
      __builtin_prefetch(wp + (long long)(k0 + 32) * Cout + n0, 0, 0);
    mma_step(As, Bs, acc, wm, wn, lane);
    __syncthreads();
  }

  // ---- epilogue per ISA 7.12.2 C/D layout: VGPR r -> M = r + 8*half, N = lane&15
  const int half = lane >> 4, mr = lane & 15;
#pragma unroll
  for (int i = 0; i < 2; ++i)
#pragma unroll
    for (int j = 0; j < 2; ++j) {
      const int colg = n0 + wn * 32 + j * 16 + mr;
      if (colg >= Cout) continue;
      const long long rowb = m0 + wm * 32 + i * 16 + half * 8;
      const float bv = bias ? bias[colg] : 0.f;
#pragma unroll
      for (int r = 0; r < 8; ++r) {
        const long long off = (rowb + r) * Cout + colg;
        float v = acc[i][j][r] + bv;
        if (resid) v += (float)resid[off];
        out[off] = (_Float16)v;
      }
    }
}

// ---------------------------------------------------------------------------
// Generic GEMM (attention): C[M,N] = alpha * A[M,K] * B[K,N], fp16 in,
// fp32 or fp16 out. Same tiling as the conv kernel.
// ---------------------------------------------------------------------------
template <bool OUTF16>
__global__ __launch_bounds__(256) void wmma_gemm_kernel(
    const _Float16* __restrict__ A, const _Float16* __restrict__ B,
    void* __restrict__ C, int M, int N, int K, float alpha) {
  __shared__ _Float16 As[128][40];
  __shared__ _Float16 Bs[32][72];
  const int tid = threadIdx.x, lane = tid & 31, wave = tid >> 5;
  const int wm = wave >> 1, wn = wave & 1;
  const long long m0 = (long long)blockIdx.x * 128;
  const int n0 = blockIdx.y * 64;

  v8f acc[2][2];
#pragma unroll
  for (int i = 0; i < 2; ++i)
#pragma unroll
    for (int j = 0; j < 2; ++j)
#pragma unroll
      for (int r = 0; r < 8; ++r) acc[i][j][r] = 0.f;

  for (int k0 = 0; k0 < K; k0 += 32) {
    for (int idx = tid; idx < 128 * 32; idx += 256) {
      const int row = idx >> 5, kk = idx & 31;
      const long long m = m0 + row;
      const int kg = k0 + kk;
      As[row][kk] = (m < M && kg < K) ? A[m * K + kg] : (_Float16)0.f;
    }
    for (int idx = tid; idx < 32 * 64; idx += 256) {
      const int kk = idx >> 6, nn = idx & 63;
      const int kg = k0 + kk, ng = n0 + nn;
      Bs[kk][nn] = (kg < K && ng < N) ? B[(long long)kg * N + ng] : (_Float16)0.f;
    }
    __syncthreads();
    if (k0 + 32 < K)
      __builtin_prefetch(B + (long long)(k0 + 32) * N + n0, 0, 0);
    mma_step(As, Bs, acc, wm, wn, lane);
    __syncthreads();
  }

  const int half = lane >> 4, mr = lane & 15;
#pragma unroll
  for (int i = 0; i < 2; ++i)
#pragma unroll
    for (int j = 0; j < 2; ++j) {
      const int colg = n0 + wn * 32 + j * 16 + mr;
      if (colg >= N) continue;
      const long long rowb = m0 + wm * 32 + i * 16 + half * 8;
#pragma unroll
      for (int r = 0; r < 8; ++r) {
        const long long row = rowb + r;
        if (row >= M) continue;
        const float v = acc[i][j][r] * alpha;
        if (OUTF16) ((_Float16*)C)[row * N + colg] = (_Float16)v;
        else        ((float*)C)[row * N + colg] = v;
      }
    }
}

// ---------------------------------------------------------------------------
// Elementwise / reduction helpers
// ---------------------------------------------------------------------------
__global__ void pack_weights_kernel(const float* __restrict__ w,
                                    _Float16* __restrict__ wp,
                                    int ci, int co, int k, long long tot) {
  const long long i = (long long)blockIdx.x * 256 + threadIdx.x;
  if (i >= tot) return;
  const int n = (int)(i % co);
  const long long kk = i / co;
  const int cc = (int)(kk % ci);
  const int r = (int)(kk / ci);
  const int kx = r % k, ky = r / k;
  wp[i] = (_Float16)w[(((long long)n * ci + cc) * k + ky) * k + kx];
}

__global__ void nchw_to_nhwc_f16_kernel(const float* __restrict__ in,
                                        _Float16* __restrict__ out,
                                        int n, int c, int h, int w, long long tot) {
  const long long idx = (long long)blockIdx.x * 256 + threadIdx.x;
  if (idx >= tot) return;
  const int cc = (int)(idx % c);
  long long t = idx / c;
  const int xx = (int)(t % w); t /= w;
  const int yy = (int)(t % h);
  const int b = (int)(t / h);
  out[idx] = (_Float16)in[(((long long)b * c + cc) * h + yy) * w + xx];
}

__global__ void nhwc_f16_to_nchw_f32_kernel(const _Float16* __restrict__ in,
                                            float* __restrict__ out,
                                            int n, int c, int h, int w, long long tot) {
  const long long idx = (long long)blockIdx.x * 256 + threadIdx.x;
  if (idx >= tot) return;
  const int xx = (int)(idx % w);
  long long t = idx / w;
  const int yy = (int)(t % h); t /= h;
  const int cc = (int)(t % c);
  const int b = (int)(t / c);
  out[idx] = (float)in[(((long long)b * h + yy) * w + xx) * c + cc];
}

__global__ void upsample2x_kernel(const _Float16* __restrict__ in,
                                  _Float16* __restrict__ out,
                                  int n, int h, int w, int c, long long tot) {
  const long long idx = (long long)blockIdx.x * 256 + threadIdx.x;
  if (idx >= tot) return;
  const int cc = (int)(idx % c);
  long long t = idx / c;
  const int x2 = (int)(t % (2 * w)); t /= (2 * w);
  const int y2 = (int)(t % (2 * h));
  const int b = (int)(t / (2 * h));
  out[idx] = in[(((long long)b * h + (y2 >> 1)) * w + (x2 >> 1)) * c + cc];
}

__global__ void transpose_f16_kernel(const _Float16* __restrict__ in,
                                     _Float16* __restrict__ out,
                                     int n, int R, int Ccol, long long tot) {
  // in [n][R][C] -> out [n][C][R]
  const long long idx = (long long)blockIdx.x * 256 + threadIdx.x;
  if (idx >= tot) return;
  const int r = (int)(idx % R);
  long long t = idx / R;
  const int cc = (int)(t % Ccol);
  const int b = (int)(t / Ccol);
  out[idx] = in[((long long)b * R + r) * Ccol + cc];
}

__global__ __launch_bounds__(256) void gn_stats_kernel(
    const _Float16* __restrict__ x, float* __restrict__ stats,
    int h, int w, int c, int cpg, float eps) {
  const int grp = blockIdx.x;           // n*32 + g
  const int b = grp / 32, g = grp % 32;
  const long long cnt = (long long)h * w * cpg;
  float s = 0.f, ss = 0.f;
  for (long long i = threadIdx.x; i < cnt; i += 256) {
    const int cc = (int)(i % cpg);
    const long long p = i / cpg;
    const int xx = (int)(p % w);
    const int yy = (int)(p / w);
    const float v = (float)x[(((long long)b * h + yy) * w + xx) * c + g * cpg + cc];
    s += v; ss += v * v;
  }
  __shared__ float r1[256], r2[256];
  r1[threadIdx.x] = s; r2[threadIdx.x] = ss;
  __syncthreads();
  for (int st = 128; st > 0; st >>= 1) {
    if (threadIdx.x < st) { r1[threadIdx.x] += r1[threadIdx.x + st]; r2[threadIdx.x] += r2[threadIdx.x + st]; }
    __syncthreads();
  }
  if (threadIdx.x == 0) {
    const float mean = r1[0] / (float)cnt;
    const float var = r2[0] / (float)cnt - mean * mean;
    stats[2 * grp] = mean;
    stats[2 * grp + 1] = rsqrtf(var + eps);
  }
}

__global__ void gn_apply_kernel(const _Float16* __restrict__ x,
                                _Float16* __restrict__ out,
                                const float* __restrict__ gamma,
                                const float* __restrict__ beta,
                                const float* __restrict__ stats,
                                long long hw, int c, int cpg, int swish, long long tot) {
  const long long idx = (long long)blockIdx.x * 256 + threadIdx.x;
  if (idx >= tot) return;
  const int cc = (int)(idx % c);
  const long long sp = idx / c;
  const int b = (int)(sp / hw);
  const int grp = b * 32 + cc / cpg;
  float v = ((float)x[idx] - stats[2 * grp]) * stats[2 * grp + 1] * gamma[cc] + beta[cc];
  if (swish) v = v / (1.f + __expf(-v));
  out[idx] = (_Float16)v;
}

__global__ __launch_bounds__(256) void softmax_kernel(
    const float* __restrict__ in, _Float16* __restrict__ out, int L, float scale) {
  const int row = blockIdx.x;
  const float* rp = in + (long long)row * L;
  __shared__ float red[256];
  float mx = -3.0e38f;
  for (int i = threadIdx.x; i < L; i += 256) mx = fmaxf(mx, rp[i] * scale);
  red[threadIdx.x] = mx; __syncthreads();
  for (int st = 128; st > 0; st >>= 1) {
    if (threadIdx.x < st) red[threadIdx.x] = fmaxf(red[threadIdx.x], red[threadIdx.x + st]);
    __syncthreads();
  }
  mx = red[0]; __syncthreads();
  float s = 0.f;
  for (int i = threadIdx.x; i < L; i += 256) s += __expf(rp[i] * scale - mx);
  red[threadIdx.x] = s; __syncthreads();
  for (int st = 128; st > 0; st >>= 1) {
    if (threadIdx.x < st) red[threadIdx.x] += red[threadIdx.x + st];
    __syncthreads();
  }
  const float inv = 1.f / red[0];
  for (int i = threadIdx.x; i < L; i += 256)
    out[(long long)row * L + i] = (_Float16)(__expf(rp[i] * scale - mx) * inv);
}

__global__ __launch_bounds__(256) void quantize_kernel(
    const _Float16* __restrict__ z, const float* __restrict__ cb,
    _Float16* __restrict__ out, int ncode, int d) {
  const long long pos = blockIdx.x;
  __shared__ float zv[16];
  __shared__ float bd[256];
  __shared__ int bi[256];
  if (threadIdx.x < d) zv[threadIdx.x] = (float)z[pos * d + threadIdx.x];
  __syncthreads();
  float best = 3.4e38f; int bidx = 0;
  for (int code = threadIdx.x; code < ncode; code += 256) {
    const float* e = cb + (long long)code * d;
    float acc = 0.f;
    for (int j = 0; j < d; ++j) { const float df = zv[j] - e[j]; acc += df * df; }
    if (acc < best) { best = acc; bidx = code; }
  }
  bd[threadIdx.x] = best; bi[threadIdx.x] = bidx;
  __syncthreads();
  for (int st = 128; st > 0; st >>= 1) {
    if (threadIdx.x < st && bd[threadIdx.x + st] < bd[threadIdx.x]) {
      bd[threadIdx.x] = bd[threadIdx.x + st]; bi[threadIdx.x] = bi[threadIdx.x + st];
    }
    __syncthreads();
  }
  if (threadIdx.x < d) out[pos * d + threadIdx.x] = (_Float16)cb[(long long)bi[0] * d + threadIdx.x];
}

// ============================================================================
// Host side: parameter walker (JAX pytree order: dicts sorted by key) + driver
// ============================================================================
namespace {

struct CP { const float* w; const float* b; int ci, co, k; };
struct NP { const float* g; const float* b; int c; };
struct RBP { CP c1, c2, nin; NP n1, n2; bool has_nin; int ci, co; };
struct ATP { CP k, q, v, proj; NP norm; int c; };
struct ELVL { std::vector<ATP> attn; CP down; bool has_down; std::vector<RBP> res; };
struct DLVL { std::vector<ATP> attn; std::vector<RBP> res; CP up; bool has_up; };
struct ENC { CP conv_in, conv_out; std::vector<ELVL> levels; RBP mid0, mid2; ATP mid1; NP norm_out; };
struct DEC { CP conv_in, conv_out; std::vector<DLVL> levels; RBP mid0, mid2; ATP mid1; NP norm_out; };

struct Walk {
  void* const* din; int idx;
  const float* next() { return (const float*)din[idx++]; }
};
static CP takeC(Walk& w, int ci, int co, int k) {      // dict {'w','b'} -> sorted: b, w
  CP p; p.ci = ci; p.co = co; p.k = k; p.b = w.next(); p.w = w.next(); return p;
}
static NP takeN(Walk& w, int c) {                      // dict {'g','b'} -> sorted: b, g
  NP p; p.c = c; p.b = w.next(); p.g = w.next(); return p;
}
static RBP takeRB(Walk& w, int ci, int co) {           // sorted: c1,c2,n1,n2,(nin)
  RBP r; r.ci = ci; r.co = co;
  r.c1 = takeC(w, ci, co, 3);
  r.c2 = takeC(w, co, co, 3);
  r.n1 = takeN(w, ci);
  r.n2 = takeN(w, co);
  r.has_nin = (ci != co);
  if (r.has_nin) r.nin = takeC(w, ci, co, 1);
  return r;
}
static ATP takeAT(Walk& w, int c) {                    // sorted: k,norm,proj,q,v
  ATP a; a.c = c;
  a.k = takeC(w, c, c, 1);
  a.norm = takeN(w, c);
  a.proj = takeC(w, c, c, 1);
  a.q = takeC(w, c, c, 1);
  a.v = takeC(w, c, c, 1);
  return a;
}
static ENC takeENC(Walk& w) {   // enc dict sorted: conv_in, conv_out, levels, mid, norm_out
  ENC e;
  e.conv_in = takeC(w, 3, 128, 3);
  e.conv_out = takeC(w, 512, 256, 3);
  const int in_mult[5] = {1, 1, 1, 2, 2};
  const int mult[5] = {1, 1, 2, 2, 4};
  for (int i = 0; i < 5; ++i) {       // level dict sorted: attn, down, res
    ELVL lv;
    const int cur = 128 * in_mult[i], bo = 128 * mult[i];
    if (i == 4) for (int j = 0; j < 2; ++j) lv.attn.push_back(takeAT(w, bo));
    lv.has_down = (i != 4);
    if (lv.has_down) lv.down = takeC(w, bo, bo, 3);
    int c = cur;
    for (int j = 0; j < 2; ++j) { lv.res.push_back(takeRB(w, c, bo)); c = bo; }
    e.levels.push_back(lv);
  }
  e.mid0 = takeRB(w, 512, 512);
  e.mid1 = takeAT(w, 512);
  e.mid2 = takeRB(w, 512, 512);
  e.norm_out = takeN(w, 512);
  return e;
}
static DEC takeDEC(Walk& w) {   // dec dict sorted: conv_in, conv_out, levels, mid, norm_out
  DEC d;
  d.conv_in = takeC(w, 256, 512, 3);
  d.conv_out = takeC(w, 128, 3, 3);
  const int mult[5] = {1, 1, 2, 2, 4};
  int cur = 512;
  for (int li = 0; li < 5; ++li) {    // level dict sorted: attn, res, up
    const int i_level = 4 - li;
    const int bo = 128 * mult[i_level];
    DLVL lv;
    if (i_level == 4) for (int j = 0; j < 3; ++j) lv.attn.push_back(takeAT(w, bo));
    int c = cur;
    for (int j = 0; j < 3; ++j) { lv.res.push_back(takeRB(w, c, bo)); c = bo; }
    cur = bo;
    lv.has_up = (i_level != 0);
    if (lv.has_up) lv.up = takeC(w, cur, cur, 3);
    d.levels.push_back(lv);
  }
  d.mid0 = takeRB(w, 512, 512);
  d.mid1 = takeAT(w, 512);
  d.mid2 = takeRB(w, 512, 512);
  d.norm_out = takeN(w, 128);
  return d;
}

struct Tens { _Float16* p; int n, h, w, c; };

struct WS {
  _Float16* big[4];
  _Float16* wpack;
  float* gnstats;
  _Float16 *ah, *aq, *ak, *akt, *av, *ao, *wsm;
  float* logits;
};

static inline long long cdiv(long long a, long long b) { return (a + b - 1) / b; }

static Tens gn_run(const Tens& x, const NP& p, bool swish, _Float16* dst,
                   WS& ws, hipStream_t s) {
  const int cpg = x.c / 32;
  gn_stats_kernel<<<x.n * 32, 256, 0, s>>>(x.p, ws.gnstats, x.h, x.w, x.c, cpg, 1e-6f);
  const long long tot = (long long)x.n * x.h * x.w * x.c;
  gn_apply_kernel<<<(unsigned)cdiv(tot, 256), 256, 0, s>>>(
      x.p, dst, p.g, p.b, ws.gnstats, (long long)x.h * x.w, x.c, cpg, swish ? 1 : 0, tot);
  return {dst, x.n, x.h, x.w, x.c};
}

static Tens conv_to(_Float16* dst, const Tens& in, const CP& cp, int stride, int pad,
                    int ho, int wo, const _Float16* resid, WS& ws, hipStream_t s) {
  const long long K = (long long)cp.k * cp.k * cp.ci;
  const long long tot = K * cp.co;
  pack_weights_kernel<<<(unsigned)cdiv(tot, 256), 256, 0, s>>>(cp.w, ws.wpack, cp.ci, cp.co, cp.k, tot);
  const long long M = (long long)in.n * ho * wo;   // always a multiple of 128 here
  dim3 grid((unsigned)(M / 128), (unsigned)cdiv(cp.co, 64));
  wmma_conv_kernel<<<grid, 256, 0, s>>>(in.p, ws.wpack, cp.b, resid, dst,
                                        in.h, in.w, cp.ci, ho, wo, cp.co, cp.k, stride, pad);
  return {dst, in.n, ho, wo, cp.co};
}

static Tens resblock_run(const Tens& X, const RBP& p, _Float16* t1, _Float16* t2,
                         _Float16* t3, WS& ws, hipStream_t s) {
  Tens g1 = gn_run(X, p.n1, true, t1, ws, s);
  Tens h1 = conv_to(t2, g1, p.c1, 1, 1, X.h, X.w, nullptr, ws, s);
  Tens g2 = gn_run(h1, p.n2, true, t1, ws, s);
  const _Float16* sc = X.p;
  if (p.has_nin) { Tens scT = conv_to(t3, X, p.nin, 1, 0, X.h, X.w, nullptr, ws, s); sc = scT.p; }
  return conv_to(t2, g2, p.c2, 1, 1, X.h, X.w, sc, ws, s);
}

static Tens attn_run(const Tens& X, const ATP& p, _Float16* outbuf, WS& ws, hipStream_t s) {
  const int hw = X.h * X.w, c = X.c, n = X.n;
  Tens g = gn_run(X, p.norm, false, ws.ah, ws, s);
  Tens q = conv_to(ws.aq, g, p.q, 1, 0, X.h, X.w, nullptr, ws, s);
  Tens k = conv_to(ws.ak, g, p.k, 1, 0, X.h, X.w, nullptr, ws, s);
  Tens v = conv_to(ws.av, g, p.v, 1, 0, X.h, X.w, nullptr, ws, s);
  (void)q; (void)k; (void)v;
  const long long tot = (long long)n * hw * c;
  transpose_f16_kernel<<<(unsigned)cdiv(tot, 256), 256, 0, s>>>(ws.ak, ws.akt, n, hw, c, tot);
  const float scale = 1.0f / sqrtf((float)c);
  for (int b = 0; b < n; ++b) {
    dim3 g1((unsigned)cdiv(hw, 128), (unsigned)cdiv(hw, 64));
    wmma_gemm_kernel<false><<<g1, 256, 0, s>>>(ws.aq + (long long)b * hw * c,
                                               ws.akt + (long long)b * c * hw,
                                               (void*)ws.logits, hw, hw, c, 1.0f);
    softmax_kernel<<<hw, 256, 0, s>>>(ws.logits, ws.wsm, hw, scale);
    dim3 g2((unsigned)cdiv(hw, 128), (unsigned)cdiv(c, 64));
    wmma_gemm_kernel<true><<<g2, 256, 0, s>>>(ws.wsm, ws.av + (long long)b * hw * c,
                                              (void*)(ws.ao + (long long)b * hw * c),
                                              hw, c, hw, 1.0f);
  }
  Tens o{ws.ao, n, X.h, X.w, c};
  return conv_to(outbuf, o, p.proj, 1, 0, X.h, X.w, X.p, ws, s);
}

}  // namespace

extern "C" void kernel_launch(void* const* d_in, const int* in_sizes, int n_in,
                              void* d_out, int out_size, void* d_ws, size_t ws_size,
                              hipStream_t stream) {
  (void)in_sizes; (void)n_in; (void)out_size; (void)ws_size;

  // ---- workspace arena ----
  char* base = (char*)d_ws;
  size_t off = 0;
  auto alloc = [&](size_t bytes) -> void* {
    void* p = base + off;
    off = (off + bytes + 255) & ~(size_t)255;
    return p;
  };
  WS ws;
  const long long BIGE = 4ll * 256 * 256 * 128;         // largest activation (elements)
  for (int i = 0; i < 4; ++i) ws.big[i] = (_Float16*)alloc((size_t)BIGE * 2);
  ws.wpack = (_Float16*)alloc((size_t)4608 * 512 * 2);  // max K*Cout
  ws.gnstats = (float*)alloc((size_t)4 * 32 * 2 * 4);
  const long long ATTE = 4ll * 256 * 512;               // hw=256 (16x16), c=512
  ws.ah = (_Float16*)alloc((size_t)ATTE * 2);
  ws.aq = (_Float16*)alloc((size_t)ATTE * 2);
  ws.ak = (_Float16*)alloc((size_t)ATTE * 2);
  ws.akt = (_Float16*)alloc((size_t)ATTE * 2);
  ws.av = (_Float16*)alloc((size_t)ATTE * 2);
  ws.ao = (_Float16*)alloc((size_t)ATTE * 2);
  ws.logits = (float*)alloc((size_t)256 * 256 * 4);
  ws.wsm = (_Float16*)alloc((size_t)256 * 256 * 2);

  // ---- parameters: d_in[0] = x, then params pytree (jax sorted-key order) ----
  const float* x_in = (const float*)d_in[0];
  Walk w{d_in, 1};
  const float* codebook = w.next();                 // 'codebook'
  DEC dec = takeDEC(w);                             // 'dec'
  ENC enc = takeENC(w);                             // 'enc'
  CP post_quant = takeC(w, 8, 256, 1);              // 'post_quant_conv'
  CP quant = takeC(w, 256, 8, 1);                   // 'quant_conv'

  // ---- buffer rotation ----
  Tens cur;
  int cb = 0;
  auto do_res = [&](const RBP& p) {
    int o[3], m = 0;
    for (int i = 0; i < 4; ++i) if (i != cb) o[m++] = i;
    cur = resblock_run(cur, p, ws.big[o[0]], ws.big[o[1]], ws.big[o[2]], ws, stream);
    cb = o[1];
  };
  auto do_attn = [&](const ATP& p) {
    const int o = (cb + 1) % 4;
    cur = attn_run(cur, p, ws.big[o], ws, stream);
    cb = o;
  };
  auto do_gn = [&](const NP& p, bool swish) {
    const int o = (cb + 1) % 4;
    cur = gn_run(cur, p, swish, ws.big[o], ws, stream);
    cb = o;
  };
  auto do_conv = [&](const CP& cp, int stride, int pad, int ho, int wo) {
    const int o = (cb + 1) % 4;
    cur = conv_to(ws.big[o], cur, cp, stride, pad, ho, wo, nullptr, ws, stream);
    cb = o;
  };

  // ---- input: fp32 NCHW -> fp16 NHWC ----
  {
    const long long tot = 4ll * 3 * 256 * 256;
    nchw_to_nhwc_f16_kernel<<<(unsigned)cdiv(tot, 256), 256, 0, stream>>>(
        x_in, ws.big[0], 4, 3, 256, 256, tot);
    cur = {ws.big[0], 4, 256, 256, 3};
    cb = 0;
  }

  // ================= encoder =================
  do_conv(enc.conv_in, 1, 1, cur.h, cur.w);
  for (int i = 0; i < 5; ++i) {
    ELVL& lv = enc.levels[i];
    for (int j = 0; j < 2; ++j) {
      do_res(lv.res[j]);
      if (!lv.attn.empty()) do_attn(lv.attn[j]);
    }
    if (lv.has_down) do_conv(lv.down, 2, 0, cur.h / 2, cur.w / 2);  // pad(0,1,0,1)+s2p0
  }
  do_res(enc.mid0);
  do_attn(enc.mid1);
  do_res(enc.mid2);
  do_gn(enc.norm_out, true);
  do_conv(enc.conv_out, 1, 1, cur.h, cur.w);        // (4,16,16,256)

  // ================= quantize =================
  do_conv(quant, 1, 0, cur.h, cur.w);               // (4,16,16,8)
  {
    const int o = (cb + 1) % 4;
    quantize_kernel<<<cur.n * cur.h * cur.w, 256, 0, stream>>>(
        cur.p, codebook, ws.big[o], 16384, 8);
    cur = {ws.big[o], cur.n, cur.h, cur.w, 8};
    cb = o;
  }
  do_conv(post_quant, 1, 0, cur.h, cur.w);          // (4,16,16,256)

  // ================= decoder =================
  do_conv(dec.conv_in, 1, 1, cur.h, cur.w);         // 512ch
  do_res(dec.mid0);
  do_attn(dec.mid1);
  do_res(dec.mid2);
  for (int li = 0; li < 5; ++li) {
    DLVL& lv = dec.levels[li];
    for (int j = 0; j < 3; ++j) {
      do_res(lv.res[j]);
      if (!lv.attn.empty()) do_attn(lv.attn[j]);
    }
    if (lv.has_up) {
      const int o = (cb + 1) % 4;
      const long long tot = (long long)cur.n * (2 * cur.h) * (2 * cur.w) * cur.c;
      upsample2x_kernel<<<(unsigned)cdiv(tot, 256), 256, 0, stream>>>(
          cur.p, ws.big[o], cur.n, cur.h, cur.w, cur.c, tot);
      cur = {ws.big[o], cur.n, 2 * cur.h, 2 * cur.w, cur.c};
      cb = o;
      do_conv(lv.up, 1, 1, cur.h, cur.w);
    }
  }
  do_gn(dec.norm_out, true);
  do_conv(dec.conv_out, 1, 1, cur.h, cur.w);        // (4,256,256,3)

  // ---- output: fp16 NHWC -> fp32 NCHW ----
  {
    const long long tot = 4ll * 3 * 256 * 256;
    nhwc_f16_to_nchw_f32_kernel<<<(unsigned)cdiv(tot, 256), 256, 0, stream>>>(
        cur.p, (float*)d_out, 4, 3, 256, 256, tot);
  }
}
